// SelfAttentionBlock_62526133895214
// MI455X (gfx1250) — matrix operand order repaired and checked by reading, not compiled, not run on hardware
//
#include <hip/hip_runtime.h>
#include <hip/hip_bf16.h>

typedef __bf16 bf16;
typedef __attribute__((ext_vector_type(16))) __bf16 v16bf;
typedef __attribute__((ext_vector_type(8)))  __bf16 v8bf;
typedef __attribute__((ext_vector_type(8)))  float  v8f;
typedef __attribute__((ext_vector_type(4)))  int    v4i;

#if defined(__AMDGCN__) && __has_builtin(__builtin_amdgcn_global_load_async_to_lds_b128) && __has_builtin(__builtin_amdgcn_s_wait_asynccnt)
#define USE_ASYNC_LDS 1
#else
#define USE_ASYNC_LDS 0
#endif

// Problem constants
constexpr int kB  = 8;
constexpr int kC  = 512;
constexpr int kS  = 1024;          // 32*32
constexpr int kBS = kB * kS;       // 8192
constexpr int kNH = 8;
constexpr int kDH = 64;

// ---------------- workspace layout (bytes) ----------------
constexpr size_t OFF_H     = 0;                                   // fp32 (BS,C) residual stream
constexpr size_t OFF_T     = OFF_H     + (size_t)kBS*kC*4;        // fp32 (BS,C) temp; also hosts Vt (bf16 8MB) during attention
constexpr size_t OFF_ABUF  = OFF_T     + (size_t)kBS*kC*4;        // bf16 (BS,C) GEMM A input
constexpr size_t OFF_OBUF  = OFF_ABUF  + (size_t)kBS*kC*2;        // bf16 (BS,C) attn output
constexpr size_t OFF_QKV   = OFF_OBUF  + (size_t)kBS*kC*2;        // bf16 (BS,3C)
constexpr size_t OFF_WCIN  = OFF_QKV   + (size_t)kBS*3*kC*2;      // bf16 weights arena (all stored n-major (N,K))
constexpr size_t OFF_QKV1  = OFF_WCIN  + (size_t)kC*kC*2;
constexpr size_t OFF_OUT1  = OFF_QKV1  + (size_t)kC*3*kC*2;
constexpr size_t OFF_QKV2  = OFF_OUT1  + (size_t)kC*kC*2;
constexpr size_t OFF_OUT2  = OFF_QKV2  + (size_t)kC*3*kC*2;
constexpr size_t OFF_G1    = OFF_OUT2  + (size_t)kC*kC*2;
constexpr size_t OFF_G2    = OFF_G1    + (size_t)kC*8*kC*2;
constexpr size_t OFF_WCOUT = OFF_G2    + (size_t)4*kC*kC*2;
constexpr size_t OFF_STATS = OFF_WCOUT + (size_t)kC*kC*2;         // fp32 256*2 GN stats
constexpr size_t OFF_SCORE = OFF_STATS + 2048;                    // bf16 64*(1024*1024) scores
constexpr size_t OFF_GBUF  = OFF_SCORE;                           // bf16 (BS,8C) -- reuses scores
constexpr size_t OFF_VBUF  = OFF_GBUF  + (size_t)kBS*8*kC*2;      // bf16 (BS,4C)
// peak ws usage ~= 222 MB

// ---------------- generic batched bf16 WMMA GEMM ----------------
// A: (M,K) row-major, lda. B: n-major (N,K) rows, ldb = row stride.
struct GemmP {
  const bf16* A; const bf16* Bmat;
  float* outF; bf16* outB;
  const float* bias; const float* res;
  long long sAo, sAi, sBo, sBi, sOo, sOi;  // outer/inner batch strides (elements)
  int nInner;
  int lda, ldb, ldo, ldr;
  int M, N, K;
  float alpha;
};

__global__ __launch_bounds__(256) void gemm_bf16_wmma(GemmP p) {
  __shared__ __align__(16) bf16 lA[2][128][40];   // [buf][m][k], pitch 40 (80B): conflict-free, 16B aligned
  __shared__ __align__(16) bf16 lB[2][128][40];   // [buf][n][k]

  const int tid  = threadIdx.x;
  const int zo   = blockIdx.z / p.nInner;
  const int zi   = blockIdx.z % p.nInner;
  const bf16* A  = p.A    + (long long)zo * p.sAo + (long long)zi * p.sAi;
  const bf16* Bm = p.Bmat + (long long)zo * p.sBo + (long long)zi * p.sBi;
  const long long obase = (long long)zo * p.sOo + (long long)zi * p.sOi;

  const int rowBase = blockIdx.x * 128;
  const int colBase = blockIdx.y * 128;
  const int lane = tid & 31, w = tid >> 5;
  const int wm = w & 3, wn = w >> 2;        // 4x2 wave grid
  const int hh = lane >> 4, l16 = lane & 15;

  v8f acc[2][4];
  #pragma unroll
  for (int r = 0; r < 2; ++r)
    #pragma unroll
    for (int c = 0; c < 4; ++c)
      #pragma unroll
      for (int e = 0; e < 8; ++e) acc[r][c][e] = 0.0f;

  const int nsteps = p.K >> 5;      // K is always a multiple of 32 here

  auto compute = [&](int buf) {
    v16bf af[2], bfr[4];
    #pragma unroll
    for (int r = 0; r < 2; ++r) {            // A frag: VGPR0-3 -> K=8h..8h+7, VGPR4-7 -> K=16+8h..
      int m = wm * 32 + r * 16 + l16;
      v8bf lo = *(const v8bf*)&lA[buf][m][8 * hh];
      v8bf hi = *(const v8bf*)&lA[buf][m][16 + 8 * hh];
      af[r] = __builtin_shufflevector(lo, hi, 0,1,2,3,4,5,6,7,8,9,10,11,12,13,14,15);
    }
    #pragma unroll
    for (int c = 0; c < 4; ++c) {            // B frag: VGPR0-3 -> K=8h..8h+7, VGPR4-7 -> K=16+8h..
      int n = wn * 64 + c * 16 + l16;
      v8bf lo = *(const v8bf*)&lB[buf][n][8 * hh];
      v8bf hi = *(const v8bf*)&lB[buf][n][16 + 8 * hh];
      bfr[c] = __builtin_shufflevector(lo, hi, 0,1,2,3,4,5,6,7,8,9,10,11,12,13,14,15);
    }
    #pragma unroll
    for (int r = 0; r < 2; ++r)
      #pragma unroll
      for (int c = 0; c < 4; ++c)
        acc[r][c] = __builtin_amdgcn_wmma_f32_16x16x32_bf16(
            false, af[r], false, bfr[c], (short)0, acc[r][c], false, false);
  };

#if USE_ASYNC_LDS
  // CDNA5 path: GLOBAL_LOAD_ASYNC_TO_LDS_B128 + ASYNCcnt, LDS double-buffered.
  typedef __attribute__((address_space(3))) v4i* as3_v4i;
  auto issueTile = [&](int s, int buf) {
    const int kBase = s << 5;
    #pragma unroll
    for (int i = 0; i < 2; ++i) {
      int idx = tid + 256 * i;
      int r = idx >> 2, cg = idx & 3;        // 512 x 16B chunks each for A and B
      __builtin_amdgcn_global_load_async_to_lds_b128(
          (v4i*)(A + (long long)(rowBase + r) * p.lda + kBase + cg * 8),
          (as3_v4i)&lA[buf][r][cg * 8], 0, 0);
      int nrow = colBase + r; if (nrow >= p.N) nrow = p.N - 1;  // clamp: masked at epilogue
      __builtin_amdgcn_global_load_async_to_lds_b128(
          (v4i*)(Bm + (long long)nrow * p.ldb + kBase + cg * 8),
          (as3_v4i)&lB[buf][r][cg * 8], 0, 0);
    }
  };
  issueTile(0, 0);
  for (int s = 0; s < nsteps; ++s) {
    const int cur = s & 1;
    __builtin_amdgcn_s_wait_asynccnt(0);     // my tile-s copies landed
    __syncthreads();                         // everyone's landed; prev compute done
    if (s + 1 < nsteps) issueTile(s + 1, cur ^ 1);
    compute(cur);
  }
#else
  // fallback: synchronous branch-free staging through registers
  v8bf aReg[2], bReg[2];
  auto loadGlobal = [&](int s) {
    const int kBase = s << 5;
    #pragma unroll
    for (int i = 0; i < 2; ++i) {
      int idx = tid + 256 * i;
      int r = idx >> 2, cg = idx & 3;
      aReg[i] = *(const v8bf*)(A + (long long)(rowBase + r) * p.lda + kBase + cg * 8);
      int nrow = colBase + r; if (nrow >= p.N) nrow = p.N - 1;  // clamp: masked at epilogue
      bReg[i] = *(const v8bf*)(Bm + (long long)nrow * p.ldb + kBase + cg * 8);
    }
  };
  auto storeLds = [&]() {
    #pragma unroll
    for (int i = 0; i < 2; ++i) {
      int idx = tid + 256 * i;
      int r = idx >> 2, cg = idx & 3;
      *(v8bf*)&lA[0][r][cg * 8] = aReg[i];
      *(v8bf*)&lB[0][r][cg * 8] = bReg[i];
    }
  };
  loadGlobal(0);
  for (int s = 0; s < nsteps; ++s) {
    __syncthreads();
    storeLds();
    __syncthreads();
    if (s + 1 < nsteps) loadGlobal(s + 1);
    compute(0);
  }
#endif

  // epilogue: alpha, bias, residual, fp32/bf16 stores
  #pragma unroll
  for (int r = 0; r < 2; ++r)
    #pragma unroll
    for (int c = 0; c < 4; ++c)
      #pragma unroll
      for (int e = 0; e < 8; ++e) {
        int m = rowBase + wm * 32 + r * 16 + e + 8 * hh;
        int n = colBase + wn * 64 + c * 16 + l16;
        if (m < p.M && n < p.N) {
          float v = acc[r][c][e] * p.alpha;
          if (p.bias) v += p.bias[n];
          if (p.res)  v += p.res[(long long)m * p.ldr + n];
          long long o = obase + (long long)m * p.ldo + n;
          if (p.outF) p.outF[o] = v;
          if (p.outB) p.outB[o] = (bf16)v;
        }
      }
}

// ---------------- GroupNorm ----------------
__global__ __launch_bounds__(256) void gn_stats_kernel(const float* __restrict__ x,
                                                       float* __restrict__ stats) {
  const int grp = blockIdx.x;                      // b*32 + g; 16ch*1024 = 16384 contiguous floats
  const float* p = x + (long long)grp * 16384;
  float s = 0.f, s2 = 0.f;
  for (int i = threadIdx.x; i < 16384; i += 256) { float v = p[i]; s += v; s2 += v * v; }
  __shared__ float r1[256], r2[256];
  r1[threadIdx.x] = s; r2[threadIdx.x] = s2; __syncthreads();
  for (int off = 128; off; off >>= 1) {
    if (threadIdx.x < off) { r1[threadIdx.x] += r1[threadIdx.x + off]; r2[threadIdx.x] += r2[threadIdx.x + off]; }
    __syncthreads();
  }
  if (threadIdx.x == 0) {
    float mu  = r1[0] * (1.0f / 16384.0f);
    float var = r2[0] * (1.0f / 16384.0f) - mu * mu;
    stats[grp * 2]     = mu;
    stats[grp * 2 + 1] = rsqrtf(var + 1e-6f);
  }
}

__global__ __launch_bounds__(256) void gn_apply_kernel(const float* __restrict__ x,
                                                       const float* __restrict__ stats,
                                                       const float* __restrict__ g,
                                                       const float* __restrict__ b,
                                                       bf16* __restrict__ out) {
  long long idx = (long long)blockIdx.x * 256 + threadIdx.x;   // (b,s,c) with c fastest
  int c = (int)(idx & (kC - 1));
  long long t = idx >> 9;
  int s = (int)(t & (kS - 1));
  int bb = (int)(t >> 10);
  int grp = bb * 32 + (c >> 4);
  float mu = stats[grp * 2], rs = stats[grp * 2 + 1];
  float v = x[(((long long)bb * kC + c) << 10) + s];
  out[idx] = (bf16)((v - mu) * rs * g[c] + b[c]);
}

// ---------------- LayerNorm (row length 512) -> bf16 ----------------
__global__ __launch_bounds__(256) void ln_kernel(const float* __restrict__ x,
                                                 const float* __restrict__ g,
                                                 const float* __restrict__ b,
                                                 bf16* __restrict__ out) {
  const int row = blockIdx.x;
  const float* xr = x + (long long)row * kC;
  __shared__ float red[256];
  float v0 = xr[threadIdx.x], v1 = xr[threadIdx.x + 256];
  red[threadIdx.x] = v0 + v1; __syncthreads();
  for (int off = 128; off; off >>= 1) { if (threadIdx.x < off) red[threadIdx.x] += red[threadIdx.x + off]; __syncthreads(); }
  float mu = red[0] * (1.0f / kC); __syncthreads();
  float d0 = v0 - mu, d1 = v1 - mu;
  red[threadIdx.x] = d0 * d0 + d1 * d1; __syncthreads();
  for (int off = 128; off; off >>= 1) { if (threadIdx.x < off) red[threadIdx.x] += red[threadIdx.x + off]; __syncthreads(); }
  float rs = rsqrtf(red[0] * (1.0f / kC) + 1e-5f);
  out[(long long)row * kC + threadIdx.x]       = (bf16)(d0 * rs * g[threadIdx.x]       + b[threadIdx.x]);
  out[(long long)row * kC + threadIdx.x + 256] = (bf16)(d1 * rs * g[threadIdx.x + 256] + b[threadIdx.x + 256]);
}

// ---------------- row softmax over 1024 bf16, in place ----------------
__global__ __launch_bounds__(256) void softmax_kernel(bf16* __restrict__ sc) {
  bf16* p = sc + (long long)blockIdx.x * 1024;
  float v[4];
  #pragma unroll
  for (int i = 0; i < 4; ++i) v[i] = (float)p[threadIdx.x + 256 * i];
  __shared__ float red[256];
  float m = fmaxf(fmaxf(v[0], v[1]), fmaxf(v[2], v[3]));
  red[threadIdx.x] = m; __syncthreads();
  for (int off = 128; off; off >>= 1) { if (threadIdx.x < off) red[threadIdx.x] = fmaxf(red[threadIdx.x], red[threadIdx.x + off]); __syncthreads(); }
  m = red[0]; __syncthreads();
  float s = 0.f;
  #pragma unroll
  for (int i = 0; i < 4; ++i) { v[i] = __expf(v[i] - m); s += v[i]; }
  red[threadIdx.x] = s; __syncthreads();
  for (int off = 128; off; off >>= 1) { if (threadIdx.x < off) red[threadIdx.x] += red[threadIdx.x + off]; __syncthreads(); }
  float inv = 1.0f / red[0];
  #pragma unroll
  for (int i = 0; i < 4; ++i) p[threadIdx.x + 256 * i] = (bf16)(v[i] * inv);
}

// ---------------- V transpose: vt[z][d][t] = V[b,t,hd,d] ----------------
__global__ __launch_bounds__(256) void vtrans_kernel(const bf16* __restrict__ qkv,
                                                     bf16* __restrict__ vt) {
  long long idx = (long long)blockIdx.x * 256 + threadIdx.x;   // over 64*64*1024, t fastest
  int t = (int)(idx & (kS - 1));
  long long zd = idx >> 10;
  int d = (int)(zd & (kDH - 1));
  int z = (int)(zd >> 6);
  int b = z >> 3, hd = z & 7;
  vt[idx] = qkv[(long long)(b * kS + t) * (3 * kC) + 2 * kC + hd * kDH + d];
}

// ---------------- GEGLU: vbuf = val * gelu_exact(gate) ----------------
__global__ __launch_bounds__(256) void geglu_kernel(const bf16* __restrict__ g4,
                                                    bf16* __restrict__ out) {
  long long idx = (long long)blockIdx.x * 256 + threadIdx.x;   // over 8192*2048
  long long m = idx >> 11; int j = (int)(idx & 2047);
  float val = (float)g4[m * 4096 + j];
  float x   = (float)g4[m * 4096 + 2048 + j];
  float ge  = 0.5f * x * (1.0f + erff(x * 0.70710678118654752f));
  out[idx] = (bf16)(val * ge);
}

// ---------------- weight convert fp32 -> bf16 (opt. transpose to (N,K)) ----------------
__global__ __launch_bounds__(256) void convw_kernel(const float* __restrict__ src,
                                                    bf16* __restrict__ dst,
                                                    int R, int Cc, int trans) {
  int idx = blockIdx.x * 256 + threadIdx.x;
  if (idx >= R * Cc) return;
  if (!trans) dst[idx] = (bf16)src[idx];
  else { int r = idx / Cc, c = idx - r * Cc; dst[c * R + r] = (bf16)src[idx]; }
}

// ---------------- final: out[b,c,s] = t[(b,s),c] + x ----------------
__global__ __launch_bounds__(256) void final_kernel(const float* __restrict__ t,
                                                    const float* __restrict__ x,
                                                    float* __restrict__ out) {
  long long idx = (long long)blockIdx.x * 256 + threadIdx.x;   // (b,c,s) with s fastest
  int s = (int)(idx & (kS - 1));
  long long bc = idx >> 10; int c = (int)(bc & (kC - 1)); int b = (int)(bc >> 9);
  out[idx] = t[((long long)b * kS + s) * kC + c] + x[idx];
}

// ---------------- host orchestration ----------------
static void run_gemm(hipStream_t st, const bf16* A, const bf16* Bm, float* oF, bf16* oB,
                     const float* bias, const float* res, int M, int N, int K,
                     int lda, int ldb, int ldo, int ldr, float alpha,
                     int nOuter, int nInner,
                     long long sAo, long long sAi, long long sBo, long long sBi,
                     long long sOo, long long sOi) {
  GemmP p;
  p.A = A; p.Bmat = Bm; p.outF = oF; p.outB = oB; p.bias = bias; p.res = res;
  p.sAo = sAo; p.sAi = sAi; p.sBo = sBo; p.sBi = sBi; p.sOo = sOo; p.sOi = sOi;
  p.nInner = nInner; p.lda = lda; p.ldb = ldb; p.ldo = ldo; p.ldr = ldr;
  p.M = M; p.N = N; p.K = K; p.alpha = alpha;
  dim3 grid(M / 128, (N + 127) / 128, nOuter * nInner);
  gemm_bf16_wmma<<<grid, 256, 0, st>>>(p);
}

extern "C" void kernel_launch(void* const* d_in, const int* in_sizes, int n_in,
                              void* d_out, int out_size, void* d_ws, size_t ws_size,
                              hipStream_t stream) {
  const float* x      = (const float*)d_in[0];
  const float* gn_g   = (const float*)d_in[1];
  const float* gn_b   = (const float*)d_in[2];
  const float* w_cin  = (const float*)d_in[3];
  const float* b_cin  = (const float*)d_in[4];
  const float* ln1_g  = (const float*)d_in[5];
  const float* ln1_b  = (const float*)d_in[6];
  const float* qkv1   = (const float*)d_in[7];
  const float* out1_w = (const float*)d_in[8];
  const float* out1_b = (const float*)d_in[9];
  const float* ln2_g  = (const float*)d_in[10];
  const float* ln2_b  = (const float*)d_in[11];
  const float* qkv2   = (const float*)d_in[12];
  const float* out2_w = (const float*)d_in[13];
  const float* out2_b = (const float*)d_in[14];
  const float* ln3_g  = (const float*)d_in[15];
  const float* ln3_b  = (const float*)d_in[16];
  const float* g1_w   = (const float*)d_in[17];
  const float* g1_b   = (const float*)d_in[18];
  const float* g2_w   = (const float*)d_in[19];
  const float* g2_b   = (const float*)d_in[20];
  const float* w_cout = (const float*)d_in[21];
  const float* b_cout = (const float*)d_in[22];
  float* out = (float*)d_out;

  char* ws = (char*)d_ws;
  float* h     = (float*)(ws + OFF_H);
  float* t     = (float*)(ws + OFF_T);
  bf16* vt     = (bf16*) (ws + OFF_T);       // Vt (z,d,t) 8MB, lives only during attention
  bf16* abuf   = (bf16*)(ws + OFF_ABUF);
  bf16* obuf   = (bf16*)(ws + OFF_OBUF);
  bf16* qkvb   = (bf16*)(ws + OFF_QKV);
  bf16* wcinN  = (bf16*)(ws + OFF_WCIN);
  bf16* qkv1N  = (bf16*)(ws + OFF_QKV1);
  bf16* out1N  = (bf16*)(ws + OFF_OUT1);
  bf16* qkv2N  = (bf16*)(ws + OFF_QKV2);
  bf16* out2N  = (bf16*)(ws + OFF_OUT2);
  bf16* g1N    = (bf16*)(ws + OFF_G1);
  bf16* g2N    = (bf16*)(ws + OFF_G2);
  bf16* wcoutN = (bf16*)(ws + OFF_WCOUT);
  float* stats = (float*)(ws + OFF_STATS);
  bf16* score  = (bf16*)(ws + OFF_SCORE);
  bf16* gbuf   = (bf16*)(ws + OFF_GBUF);
  bf16* vbuf   = (bf16*)(ws + OFF_VBUF);

  auto cw = [&](const float* s, bf16* d, int R, int Cc, int tr) {
    convw_kernel<<<(R * Cc + 255) / 256, 256, 0, stream>>>(s, d, R, Cc, tr);
  };
  // all weights stored n-major (N,K)
  cw(w_cin,  wcinN,  kC, kC, 0);        // (o,c) is already (N,K)
  cw(qkv1,   qkv1N,  kC, 3 * kC, 1);    // -> (3C, C)
  cw(out1_w, out1N,  kC, kC, 1);
  cw(qkv2,   qkv2N,  kC, 3 * kC, 1);
  cw(out2_w, out2N,  kC, kC, 1);
  cw(g1_w,   g1N,    kC, 8 * kC, 1);    // -> (8C, C)
  cw(g2_w,   g2N,    4 * kC, kC, 1);    // -> (C, 4C)
  cw(w_cout, wcoutN, kC, kC, 0);        // (o,c) is already (N,K)

  // GroupNorm + transpose to (BS, C) bf16
  gn_stats_kernel<<<kB * 32, 256, 0, stream>>>(x, stats);
  gn_apply_kernel<<<(kB * kS * kC) / 256, 256, 0, stream>>>(x, stats, gn_g, gn_b, abuf);

  // conv_in: h = abuf @ WcinN^T + b_cin   (fp32 residual stream)
  run_gemm(stream, abuf, wcinN, h, nullptr, b_cin, nullptr,
           kBS, kC, kC, kC, kC, kC, 0, 1.0f, 1, 1, 0,0,0,0,0,0);

  const long long sBH = (long long)kS * 3 * kC;   // per-batch stride inside qkv buffer
  const float* lng[2] = { ln1_g, ln2_g };
  const float* lnb[2] = { ln1_b, ln2_b };
  const bf16*  wq[2]  = { qkv1N, qkv2N };
  const bf16*  wo[2]  = { out1N, out2N };
  const float* ob[2]  = { out1_b, out2_b };

  for (int layer = 0; layer < 2; ++layer) {
    ln_kernel<<<kBS, 256, 0, stream>>>(h, lng[layer], lnb[layer], abuf);
    // qkv = ln(h) @ Wqkv   -> bf16 (BS, 3C)
    run_gemm(stream, abuf, wq[layer], nullptr, qkvb, nullptr, nullptr,
             kBS, 3 * kC, kC, kC, kC, 3 * kC, 0, 1.0f, 1, 1, 0,0,0,0,0,0);
    // scores = Q @ K^T / 8  (batch = 8 batches x 8 heads); K rows are naturally n-major
    run_gemm(stream, qkvb, qkvb + kC, nullptr, score, nullptr, nullptr,
             kS, kS, kDH, 3 * kC, 3 * kC, kS, 0, 0.125f,
             kB, kNH, sBH, kDH, sBH, kDH,
             (long long)kNH * kS * kS, (long long)kS * kS);
    softmax_kernel<<<kB * kNH * kS, 256, 0, stream>>>(score);
    // V -> Vt (z, d, t) so the AV GEMM's B operand is n-major
    vtrans_kernel<<<(kB * kNH * kDH * kS) / 256, 256, 0, stream>>>(qkvb, vt);
    // O = P @ V  -> obuf (BS, C) with head-column offsets
    run_gemm(stream, score, vt, nullptr, obuf, nullptr, nullptr,
             kS, kDH, kS, kS, kS, kC, 0, 1.0f,
             kB, kNH, (long long)kNH * kS * kS, (long long)kS * kS,
             (long long)kNH * kDH * kS, (long long)kDH * kS,
             (long long)kS * kC, kDH);
    // h = O @ Wout + b + h
    run_gemm(stream, obuf, wo[layer], h, nullptr, ob[layer], h,
             kBS, kC, kC, kC, kC, kC, kC, 1.0f, 1, 1, 0,0,0,0,0,0);
  }

  // FFN with GEGLU
  ln_kernel<<<kBS, 256, 0, stream>>>(h, ln3_g, ln3_b, abuf);
  run_gemm(stream, abuf, g1N, nullptr, gbuf, g1_b, nullptr,
           kBS, 8 * kC, kC, kC, kC, 8 * kC, 0, 1.0f, 1, 1, 0,0,0,0,0,0);
  geglu_kernel<<<(kBS * 4 * kC) / 256, 256, 0, stream>>>(gbuf, vbuf);
  // h = vbuf @ G2 + b + h ; also produce bf16 copy for conv_out
  run_gemm(stream, vbuf, g2N, h, abuf, g2_b, h,
           kBS, kC, 4 * kC, 4 * kC, 4 * kC, kC, kC, 1.0f, 1, 1, 0,0,0,0,0,0);
  // conv_out: t = h_bf16 @ WcoutN^T + b_cout
  run_gemm(stream, abuf, wcoutN, t, nullptr, b_cout, nullptr,
           kBS, kC, kC, kC, kC, kC, 0, 1.0f, 1, 1, 0,0,0,0,0,0);
  // transpose back to (B,C,H,W) and add long residual
  final_kernel<<<(kB * kC * kS) / 256, 256, 0, stream>>>(t, x, out);
}
